// NER_LSTM_CRF_11072425689419
// MI455X (gfx1250) — matrix operand order repaired and checked by reading, not compiled, not run on hardware
//
#include <hip/hip_runtime.h>

// ---------------------------------------------------------------------------
// BiLSTM-CRF for MI455X (gfx1250, wave32, WMMA).
//   k_embed      : gather + f32->bf16 cast of embeddings
//   k_castw      : weights -> bf16 (w_out zero-padded 17->32 rows)
//   k_input_proj : XP = E @ W_ih^T + b   (WMMA bf16, async-LDS staging)
//   k_lstm       : persistent WG/direction, h,c in LDS, WMMA per timestep
//   k_emis       : EM = [h_f|h_b] @ W_out^T + b_out (WMMA, TDM staging)
//   k_viterbi    : wave-per-batch forward max-product + backtrack
// ---------------------------------------------------------------------------

typedef __bf16 v16bf __attribute__((ext_vector_type(16)));
typedef float  v8f   __attribute__((ext_vector_type(8)));
typedef unsigned int u32x4 __attribute__((ext_vector_type(4)));
typedef int          i32x4 __attribute__((ext_vector_type(4)));
typedef int          i32x8 __attribute__((ext_vector_type(8)));

#define DEV static __device__ __forceinline__

constexpr int Bn    = 32;
constexpr int Tn    = 512;
constexpr int EMBn  = 256;
constexpr int HALFn = 256;
constexpr int Gn    = 1024;   // 4*HALF
constexpr int HIDn  = 512;
constexpr int TAGSn = 17;
constexpr int Mrows = Bn * Tn;   // 16384

DEV float sigm(float x)   { return 1.0f / (1.0f + __expf(-x)); }
DEV float tanh_f(float x) { return 2.0f / (1.0f + __expf(-2.0f * x)) - 1.0f; }

DEV v16bf join16(uint4 lo, uint4 hi) {
  union { uint4 u[2]; v16bf v; } u;
  u.u[0] = lo; u.u[1] = hi;
  return u.v;
}

// A-fragment (16x32 bf16, row-major source): lane L covers M=L%16.
// element i -> K = kb + 16*(L/16) + i
DEV v16bf frag_a(const __bf16* row, int kb, int hl) {
  uint4 lo = *(const uint4*)(row + kb + 16 * hl);
  uint4 hi = *(const uint4*)(row + kb + 16 * hl + 8);
  return join16(lo, hi);
}

// B-fragment (32x16 bf16): lane L covers N=L%16; element i -> K = kb + 16*(L/16)+i.
// With W stored row-major [N][K] this is 32 contiguous bytes of row N.
DEV v16bf frag_b(const __bf16* Wrow, int kb, int hl) {
  const uint4* p = (const uint4*)(Wrow + kb + 16 * hl);
  return join16(p[0], p[1]);
}

DEV v8f wmma_bf16(v16bf a, v16bf b, v8f c) {
  return __builtin_amdgcn_wmma_f32_16x16x32_bf16(false, a, false, b, (short)0, c,
                                                 false, false);
}

// CDNA5 async global->LDS copy (ASYNCcnt-tracked). lds_off is the raw LDS byte
// offset (single __shared__ object per kernel -> frame offset 0). lds_base is
// passed (unused in the template) so the LDS object ESCAPES: without it the
// compiler proves the LDS is never written and folds the ds_loads to undef.
DEV void async_g2l_b128(unsigned lds_off, const void* gaddr,
                        const void* lds_base) {
  asm volatile("global_load_async_to_lds_b128 %0, %1, off"
               :: "v"(lds_off), "v"((unsigned long long)gaddr),
                  "v"((unsigned long long)lds_base)
               : "memory");
}
DEV void wait_async0() { asm volatile("s_wait_asynccnt 0" ::: "memory"); }

// Escape + ordering fence for an LDS object written behind the compiler's back
// (TDM descriptor path references LDS only by integer address).
DEV void lds_escape_fence(const void* lds_base) {
  asm volatile("" :: "v"((unsigned long long)lds_base) : "memory");
}

// ---------------------------------------------------------------------------
__global__ __launch_bounds__(256) void k_embed(const int* __restrict__ x,
                                               const float* __restrict__ emb,
                                               __bf16* __restrict__ E) {
  int m = blockIdx.x;                 // b*T + t
  int tok = x[m];
  int i = threadIdx.x;                // 256 = EMB
  E[(size_t)m * EMBn + i] = (__bf16)emb[(size_t)tok * EMBn + i];
}

// ---------------------------------------------------------------------------
__global__ __launch_bounds__(256) void k_castw(
    const float* __restrict__ wfi, const float* __restrict__ wfh,
    const float* __restrict__ wbi, const float* __restrict__ wbh,
    const float* __restrict__ wout,
    __bf16* __restrict__ Wfi, __bf16* __restrict__ Wfh,
    __bf16* __restrict__ Wbi, __bf16* __restrict__ Wbh,
    __bf16* __restrict__ Wout) {
  const int nW = Gn * EMBn;  // 262144 per matrix
  for (int i = blockIdx.x * blockDim.x + threadIdx.x; i < nW;
       i += gridDim.x * blockDim.x) {
    Wfi[i] = (__bf16)wfi[i];
    Wfh[i] = (__bf16)wfh[i];
    Wbi[i] = (__bf16)wbi[i];
    Wbh[i] = (__bf16)wbh[i];
  }
  for (int i = blockIdx.x * blockDim.x + threadIdx.x; i < 32 * HIDn;
       i += gridDim.x * blockDim.x) {
    int r = i >> 9;  // /512
    Wout[i] = (r < TAGSn) ? (__bf16)wout[i] : (__bf16)0.0f;
  }
}

// ---------------------------------------------------------------------------
// XP[m][g] = sum_e E[m][e] * W[g][e] + bias[g]; grid=(M/16, 8, 2), block=256.
__global__ __launch_bounds__(256) void k_input_proj(
    const __bf16* __restrict__ E,
    const __bf16* __restrict__ Wf, const __bf16* __restrict__ Wb,
    const float* __restrict__ bf_, const float* __restrict__ bb_,
    float* __restrict__ XPf, float* __restrict__ XPb) {
  __shared__ __align__(16) __bf16 shA[16 * EMBn];  // 8 KB, 16 rows of A

  const int mbase = blockIdx.x * 16;
  const int dir   = blockIdx.z;
  const __bf16* W    = dir ? Wb : Wf;
  const float*  bias = dir ? bb_ : bf_;
  float*        XP   = dir ? XPb : XPf;

  // Stage A tile (16 x 256 bf16 = 8KB) with async LDS DMA: 512 x 16B chunks.
  for (int it = 0; it < 2; ++it) {
    int idx = it * 256 + threadIdx.x;
    int row = idx >> 5;      // 32 chunks per 512B row
    int col = idx & 31;
    const uint4* gp = (const uint4*)(E + (size_t)(mbase + row) * EMBn) + col;
    async_g2l_b128((unsigned)(idx * 16), gp, shA);
  }
  wait_async0();
  __syncthreads();

  const int wave = threadIdx.x >> 5;
  const int lane = threadIdx.x & 31;
  const int hl = lane >> 4, ln = lane & 15;
  const int nbase = blockIdx.y * 128 + wave * 16;

  v8f acc = {};
  const __bf16* arow = shA + ln * EMBn;
#pragma unroll
  for (int ks = 0; ks < 8; ++ks) {
    int kb = ks * 32;
    v16bf a = frag_a(arow, kb, hl);
    v16bf b = frag_b(W + (size_t)(nbase + ln) * EMBn, kb, hl);
    acc = wmma_bf16(a, b, acc);
  }
  float bv = bias[nbase + ln];
#pragma unroll
  for (int r = 0; r < 8; ++r) {
    int m = mbase + r + 8 * hl;     // C/D layout: M = r + 8*(lane/16)
    XP[(size_t)m * Gn + nbase + ln] = acc[r] + bv;
  }
}

// ---------------------------------------------------------------------------
// Persistent recurrence. grid=2 (dir), block=1024 (32 waves).
// wave = (mtile in {0,1}) x (jtile in 0..15); each wave owns the 4 gate
// quadrants (i,f,g,o) for its (batch-half, 16-wide j slice).
__global__ __launch_bounds__(1024) void k_lstm(
    const float* __restrict__ XPf, const float* __restrict__ XPb,
    const __bf16* __restrict__ Whf, const __bf16* __restrict__ Whb,
    __bf16* __restrict__ H) {
  __shared__ __align__(16) __bf16 hbuf[2][Bn][HALFn];  // 32 KB (double buffer)
  __shared__ float cbuf[Bn][HALFn];                    // 32 KB

  const int dir = blockIdx.x;
  const float*  XP = dir ? XPb : XPf;
  const __bf16* Wh = dir ? Whb : Whf;

  for (int i = threadIdx.x; i < 2 * Bn * HALFn; i += 1024)
    ((__bf16*)hbuf)[i] = (__bf16)0.0f;
  for (int i = threadIdx.x; i < Bn * HALFn; i += 1024)
    ((float*)cbuf)[i] = 0.0f;
  __syncthreads();

  const int wave = threadIdx.x >> 5;
  const int lane = threadIdx.x & 31;
  const int hl = lane >> 4, ln = lane & 15;
  const int mtile = wave & 1;          // batch rows [16*mtile, 16*mtile+16)
  const int jb = (wave >> 1) * 16;     // j slice within HALF
  const int j = jb + ln;

  for (int t = 0; t < Tn; ++t) {
    const int cur = t & 1, nxt = cur ^ 1;
    const int tt = dir ? (Tn - 1 - t) : t;

    v8f ai = {}, af = {}, ag = {}, ao = {};
    const __bf16* arow = &hbuf[cur][mtile * 16 + ln][0];
#pragma unroll
    for (int ks = 0; ks < 8; ++ks) {
      int kb = ks * 32;
      v16bf a = frag_a(arow, kb, hl);
      const __bf16* w0 = Wh + (size_t)(0 * HALFn + jb + ln) * HALFn;
      const __bf16* w1 = Wh + (size_t)(1 * HALFn + jb + ln) * HALFn;
      const __bf16* w2 = Wh + (size_t)(2 * HALFn + jb + ln) * HALFn;
      const __bf16* w3 = Wh + (size_t)(3 * HALFn + jb + ln) * HALFn;
      __builtin_prefetch(w0 + kb + 32, 0, 1);   // global_prefetch next k-slice
      ai = wmma_bf16(a, frag_b(w0, kb, hl), ai);
      af = wmma_bf16(a, frag_b(w1, kb, hl), af);
      ag = wmma_bf16(a, frag_b(w2, kb, hl), ag);
      ao = wmma_bf16(a, frag_b(w3, kb, hl), ao);
    }

#pragma unroll
    for (int r = 0; r < 8; ++r) {
      int brow = mtile * 16 + r + 8 * hl;             // batch index 0..31
      size_t xrow = ((size_t)brow * Tn + tt) * Gn;
      float gi = ai[r] + XP[xrow + 0 * HALFn + j];
      float gf = af[r] + XP[xrow + 1 * HALFn + j];
      float gg = ag[r] + XP[xrow + 2 * HALFn + j];
      float go = ao[r] + XP[xrow + 3 * HALFn + j];
      float iv = sigm(gi), fv = sigm(gf);
      float gv = tanh_f(gg), ov = sigm(go);
      float cold = cbuf[brow][j];
      float cnew = fv * cold + iv * gv;
      cbuf[brow][j] = cnew;
      float hv = ov * tanh_f(cnew);
      hbuf[nxt][brow][j] = (__bf16)hv;
      H[((size_t)brow * Tn + tt) * HIDn + dir * HALFn + j] = (__bf16)hv;
    }
    __syncthreads();
  }
}

// ---------------------------------------------------------------------------
// EM[m][k] = (sum_h H[m][h]*Wout[k][h] + b_out[k]) * mask[m]; N padded to 32.
// grid = M/64, block = 256 (8 waves = 4 mtiles x 2 ntiles).
// A-tile staged by the Tensor Data Mover: one tensor_load_to_lds issued by
// wave 0 copies 64 rows x 1024B (2-D D#: data_size=8B, tile 128x64 units,
// tensor_dim0_stride=128 units = 1024B row pitch) into LDS offset 0.
__global__ __launch_bounds__(256) void k_emis(
    const __bf16* __restrict__ H, const __bf16* __restrict__ Wout,
    const float* __restrict__ bout, const int* __restrict__ mask,
    float* __restrict__ EM) {
  __shared__ __align__(16) __bf16 shH[64 * HIDn];   // 64 KB

  const int mbase = blockIdx.x * 64;

  if (threadIdx.x < 32) {   // one wave issues the TDM descriptor
    unsigned long long ga =
        (unsigned long long)(const void*)(H + (size_t)mbase * HIDn);
    // D# group 0: count=1 | lds_addr=0 | global_addr | type=2
    u32x4 g0 = { 1u,
                 0u,
                 (unsigned)(ga & 0xffffffffu),
                 (unsigned)(((ga >> 32) & 0x01ffffffu) | (2u << 30)) };
    // D# group 1: data_size=3(8B); tensor_dim0=128; tensor_dim1=16384;
    // tile_dim0=128; tile_dim1=64; tensor_dim0_stride=128.
    i32x8 g1 = { 0x00030000,          // wg_mask=0, data_size=3
                 0x00800000,          // tensor_dim0[15:0]=128 << 16
                 0x40000000,          // tensor_dim1[15:0]=16384 << 16
                 0x00800000,          // tile_dim0=128 << 16
                 0x00000040,          // tile_dim1=64, tile_dim2=0
                 128,                 // tensor_dim0_stride lo32
                 0, 0 };
    i32x4 gz = { 0, 0, 0, 0 };
#if __clang_major__ >= 23
    i32x8 gz8 = { 0, 0, 0, 0, 0, 0, 0, 0 };
    __builtin_amdgcn_tensor_load_to_lds(g0, g1, gz, gz, gz8, 0);
#else
    __builtin_amdgcn_tensor_load_to_lds(g0, g1, gz, gz, 0);
#endif
    __builtin_amdgcn_s_wait_tensorcnt(0);
  }
  lds_escape_fence(shH);   // shH written behind the compiler's back
  __syncthreads();

  const int wave = threadIdx.x >> 5;
  const int lane = threadIdx.x & 31;
  const int hl = lane >> 4, ln = lane & 15;
  const int mt = wave >> 1;            // local m tile 0..3
  const int nbase = (wave & 1) * 16;   // tag tile

  v8f acc = {};
  const __bf16* arow = shH + (mt * 16 + ln) * HIDn;
#pragma unroll
  for (int ks = 0; ks < 16; ++ks) {
    int kb = ks * 32;
    v16bf a = frag_a(arow, kb, hl);
    v16bf b = frag_b(Wout + (size_t)(nbase + ln) * HIDn, kb, hl);
    acc = wmma_bf16(a, b, acc);
  }
  int n = nbase + ln;
  if (n < TAGSn) {
    float bv = bout[n];
#pragma unroll
    for (int r = 0; r < 8; ++r) {
      int m = mbase + mt * 16 + r + 8 * hl;
      EM[(size_t)m * 32 + n] = (acc[r] + bv) * (float)mask[m];
    }
  }
}

// ---------------------------------------------------------------------------
// Viterbi: one wave per batch row; lane j<17 owns tag j.
__global__ __launch_bounds__(1024) void k_viterbi(
    const float* __restrict__ EM, const int* __restrict__ mask,
    const float* __restrict__ trans, const float* __restrict__ startt,
    const float* __restrict__ endt, unsigned char* __restrict__ BP,
    int* __restrict__ tags_out, float* __restrict__ score_out) {
  __shared__ float ssc[Bn][32];

  const int b = threadIdx.x >> 5;
  const int lane = threadIdx.x & 31;

  ssc[b][lane] = (lane < TAGSn)
                     ? startt[lane] + EM[((size_t)b * Tn) * 32 + lane]
                     : -3.0e38f;
  __syncthreads();

  for (int t = 1; t < Tn; ++t) {
    float ns = -3.0e38f;
    if (lane < TAGSn) {
      float best = -3.0e38f;
      int arg = 0;
      for (int i = 0; i < TAGSn; ++i) {
        float v = ssc[b][i] + trans[i * TAGSn + lane];
        if (v > best) { best = v; arg = i; }
      }
      ns = best + EM[((size_t)b * Tn + t) * 32 + lane];
      if (!(mask[b * Tn + t] > 0)) ns = ssc[b][lane];
      BP[((size_t)(t - 1) * Bn + b) * 32 + lane] = (unsigned char)arg;
    }
    __syncthreads();
    if (lane < TAGSn) ssc[b][lane] = ns;
    __syncthreads();
  }

  if (lane < TAGSn) ssc[b][lane] += endt[lane];
  __syncthreads();

  if (lane == 0) {
    float best = -3.0e38f;
    int last = 0;
    for (int i = 0; i < TAGSn; ++i) {
      float v = ssc[b][i];
      if (v > best) { best = v; last = i; }
    }
    score_out[b] = best;
    int tag = last;
    tags_out[b * Tn + (Tn - 1)] = tag;
    for (int s = Tn - 2; s >= 0; --s) {
      int prev = BP[((size_t)s * Bn + b) * 32 + tag];
      if (mask[b * Tn + s + 1] > 0) tag = prev;
      tags_out[b * Tn + s] = tag;
    }
  }
}

// ---------------------------------------------------------------------------
extern "C" void kernel_launch(void* const* d_in, const int* in_sizes, int n_in,
                              void* d_out, int out_size, void* d_ws,
                              size_t ws_size, hipStream_t stream) {
  (void)in_sizes; (void)n_in; (void)out_size; (void)ws_size;

  const int*   x      = (const int*)d_in[0];
  const int*   mask   = (const int*)d_in[1];
  const float* emb    = (const float*)d_in[2];
  const float* w_ih_f = (const float*)d_in[3];
  const float* w_hh_f = (const float*)d_in[4];
  const float* b_f    = (const float*)d_in[5];
  const float* w_ih_b = (const float*)d_in[6];
  const float* w_hh_b = (const float*)d_in[7];
  const float* b_b    = (const float*)d_in[8];
  const float* w_out  = (const float*)d_in[9];
  const float* b_out  = (const float*)d_in[10];
  const float* trans  = (const float*)d_in[11];
  const float* startt = (const float*)d_in[12];
  const float* endt   = (const float*)d_in[13];

  char* w = (char*)d_ws;
  auto carve = [&](size_t bytes) {
    char* p = w;
    w += (bytes + 255) & ~(size_t)255;
    return p;
  };

  __bf16* E    = (__bf16*)carve((size_t)Mrows * EMBn * 2);      // 8 MB
  __bf16* Wfi  = (__bf16*)carve((size_t)Gn * EMBn * 2);         // 512 KB
  __bf16* Wfh  = (__bf16*)carve((size_t)Gn * HALFn * 2);
  __bf16* Wbi  = (__bf16*)carve((size_t)Gn * EMBn * 2);
  __bf16* Wbh  = (__bf16*)carve((size_t)Gn * HALFn * 2);
  __bf16* Wout = (__bf16*)carve((size_t)32 * HIDn * 2);         // 32 KB
  float*  XPf  = (float*)carve((size_t)Mrows * Gn * 4);         // 64 MB
  float*  XPb  = (float*)carve((size_t)Mrows * Gn * 4);         // 64 MB
  __bf16* H    = (__bf16*)carve((size_t)Mrows * HIDn * 2);      // 16 MB
  float*  EM   = (float*)carve((size_t)Mrows * 32 * 4);         // 2 MB
  unsigned char* BP = (unsigned char*)carve((size_t)(Tn - 1) * Bn * 32);

  int*   tags_out  = (int*)d_out;
  float* score_out = (float*)d_out + Mrows;

  k_embed<<<Mrows, 256, 0, stream>>>(x, emb, E);
  k_castw<<<256, 256, 0, stream>>>(w_ih_f, w_hh_f, w_ih_b, w_hh_b, w_out,
                                   Wfi, Wfh, Wbi, Wbh, Wout);
  k_input_proj<<<dim3(Mrows / 16, 8, 2), 256, 0, stream>>>(E, Wfi, Wbi, b_f,
                                                           b_b, XPf, XPb);
  k_lstm<<<2, 1024, 0, stream>>>(XPf, XPb, Wfh, Wbh, H);
  k_emis<<<Mrows / 64, 256, 0, stream>>>(H, Wout, b_out, mask, EM);
  k_viterbi<<<1, 1024, 0, stream>>>(EM, mask, trans, startt, endt, BP,
                                    tags_out, score_out);
}
